// NeuralLongTermMemory_32564442038706
// MI455X (gfx1250) — matrix operand order repaired and checked by reading, not compiled, not run on hardware
//
#include <hip/hip_runtime.h>
#include <cstdint>
#include <cstddef>

// ---------------- problem dims ----------------
#define BATCH 4
#define SEQ   4096
#define DIMN  1024
#define NTOK  (BATCH*SEQ)                       // 16384 tokens

static const long long ND_ELEMS = (long long)NTOK * DIMN;   // 16,777,216
static const long long DD_ELEMS = (long long)DIMN * DIMN;   // 1,048,576

// ---------------- types / helpers ----------------
typedef __bf16 bf16;
typedef __attribute__((ext_vector_type(16))) __bf16 v16bf;
typedef __attribute__((ext_vector_type(8)))  float  v8f;

#define DEVFN __device__ __forceinline__

DEVFN bf16 f2bf(float f) {
  unsigned u = __builtin_bit_cast(unsigned, f);
  u += 0x7FFFu + ((u >> 16) & 1u);              // round-to-nearest-even
  unsigned short h = (unsigned short)(u >> 16);
  return __builtin_bit_cast(bf16, h);
}
DEVFN float bf2f(bf16 b) {
  unsigned short h = __builtin_bit_cast(unsigned short, b);
  unsigned u = ((unsigned)h) << 16;
  return __builtin_bit_cast(float, u);
}
DEVFN float fsigmoid(float x) { return 1.0f / (1.0f + __expf(-x)); }
DEVFN float fsilu(float x)    { return x * fsigmoid(x); }
DEVFN float fsilu_deriv(float x) { float s = fsigmoid(x); return s * (1.0f + x * (1.0f - s)); }

// ---------------- GEMM tiling ----------------
// Block tile 128x128, K-tile 64; 256 threads = 8 wave32; each wave = 32(M) x 64(N)
// => 2x4 grid of 16x16 WMMA tiles per wave, v_wmma_f32_16x16x32_bf16.
#define TBM 128
#define TBN 128
#define TBK 64
#define LDK 72          // padded LDS row stride in bf16 elements (144B, 16B-aligned)
#define TILE_ELEMS (TBM * LDK)
#define TILE_BYTES (TILE_ELEMS * 2)

DEVFN v8f wmma_bf16(v16bf a, v16bf b, v8f c) {
  // D = A(16x32 bf16) x B(32x16 bf16) + C(16x16 f32)
  return __builtin_amdgcn_wmma_f32_16x16x32_bf16(false, a, false, b, (short)0, c,
                                                 false, false);
}

// Low 32 bits of a generic pointer to __shared__ data == LDS byte offset
// (flat->LDS aperture mapping uses addr[31:0] directly; see ISA 10.2).
DEVFN unsigned lds_addr_of(const void* p) {
  return (unsigned)(uintptr_t)p;
}

// A-fragment (16x32, MxK) from an LDS tile laid out [row][k] with stride LDK.
// Lane layout: lanes 0-15 M=lane K=0..15; lanes 16-31 K=16..31 (16 contiguous K per lane).
DEVFN v16bf load_frag_a(const bf16* __restrict__ tile, int row0, int k0, int lane) {
  const int m  = row0 + (lane & 15);
  const int kh = (lane >> 4) * 16;
  const bf16* p = tile + m * LDK + k0 + kh;
  v16bf r;
#pragma unroll
  for (int i = 0; i < 8; ++i) r[i] = p[i];
#pragma unroll
  for (int i = 0; i < 8; ++i) r[8 + i] = p[8 + i];
  return r;
}

// B-fragment (32x16, KxN) where Bmat[k][n] = tile[n][k] (tile rows are N, cols are K).
// Lane layout: lanes 0-15 n=lane K=0..15; lanes 16-31 K=16..31 (16 contiguous K per lane).
DEVFN v16bf load_frag_b(const bf16* __restrict__ tile, int col0, int k0, int lane) {
  const int n  = col0 + (lane & 15);
  const int kh = (lane >> 4) * 16;
  const bf16* p = tile + n * LDK + k0 + kh;
  v16bf r;
#pragma unroll
  for (int i = 0; i < 16; ++i) r[i] = p[i];
  return r;
}

// Async-stage a [128 rows x 64 K] row-major tile straight into LDS (no VGPR bounce).
// Each thread issues 4x GLOBAL_LOAD_ASYNC_TO_LDS_B128 (tracked by ASYNCcnt).
DEVFN void stage_async_nt(const bf16* __restrict__ g, int ld, int r0, int k0,
                          unsigned ldsb, int tid) {
#pragma unroll
  for (int i = 0; i < 4; ++i) {
    int c   = tid + i * 256;        // 1024 chunks of 8 bf16
    int row = c >> 3;
    int col = (c & 7) * 8;
    const bf16* ga = g + (size_t)(r0 + row) * ld + k0 + col;
    unsigned la = ldsb + (unsigned)(row * LDK + col) * 2u;
    asm volatile("global_load_async_to_lds_b128 %0, %1, off"
                 :: "v"(la), "v"(ga)
                 : "memory");
  }
}

DEVFN void wait_async_le8()  { asm volatile("s_wait_asynccnt 0x8" ::: "memory"); }
DEVFN void wait_async_zero() { asm volatile("s_wait_asynccnt 0x0" ::: "memory"); }

// Synchronous transposed staging for A^T*B GEMMs: global G[S, D], tile s in
// [s0,s0+64), d in [d0,d0+128); LDS result sm[d][s] so K (=s) is contiguous.
DEVFN void stage_tile_tn(const bf16* __restrict__ g, int ld, int s0, int d0,
                         bf16* __restrict__ sm, int tid) {
#pragma unroll
  for (int i = 0; i < 4; ++i) {
    int c  = tid + i * 256;         // 1024 chunks of 8 bf16 along d
    int s  = c >> 4;                // 16 chunks per s-row (128/8)
    int dc = (c & 15) * 8;
    uint4 v = *reinterpret_cast<const uint4*>(g + (size_t)(s0 + s) * ld + d0 + dc);
    const bf16* pv = reinterpret_cast<const bf16*>(&v);
#pragma unroll
    for (int e = 0; e < 8; ++e) sm[(dc + e) * LDK + s] = pv[e];
  }
}

// ---------------- fused epilogue modes ----------------
enum {
  MODE_F32 = 0,        // Cf = acc
  MODE_BF16 = 1,       // Cb = bf16(acc)
  MODE_SILU = 2,       // Cb = bf16(silu(acc))
  MODE_SILU_DUAL = 3,  // Cb2 = bf16(acc), Cb = bf16(silu(acc))   (pre1 + a1)
  MODE_ERR = 4,        // Cb = bf16(scale * clip(acc - auxb, +-5)) (d2)
  MODE_DSILU = 5       // Cb = bf16(acc * silu'(auxb))             (d1)
};

// C[M,N] = A[M,K] @ B[N,K]^T  (both operands row-major, K contiguous)
// Double-buffered: async fill of tile t+1 overlaps WMMA on tile t.
template <int MODE>
__global__ __launch_bounds__(256) void gemm_bf16_nt(
    const bf16* __restrict__ A, const bf16* __restrict__ B,
    float* __restrict__ Cf, bf16* __restrict__ Cb, bf16* __restrict__ Cb2,
    const bf16* __restrict__ auxb, int M, int N, int K, float scale) {
  const int tid  = threadIdx.x;
  const int lane = tid & 31;
  const int wave = tid >> 5;
  const int wm   = wave & 3;          // 4 wave-rows  -> 32 M each
  const int wn   = wave >> 2;         // 2 wave-cols  -> 64 N each
  const int bm   = blockIdx.y * TBM;
  const int bn   = blockIdx.x * TBN;

  // [2 buffers][A tile | B tile] = 72 KB of the 320 KB WGP LDS
  __shared__ __align__(16) bf16 smem[2 * 2 * TILE_ELEMS];
  const unsigned smbase = lds_addr_of(smem);

  v8f acc[2][4];
#pragma unroll
  for (int mi = 0; mi < 2; ++mi)
#pragma unroll
    for (int ni = 0; ni < 4; ++ni)
#pragma unroll
      for (int r = 0; r < 8; ++r) acc[mi][ni][r] = 0.0f;

  // prologue: fill buffer 0 asynchronously
  stage_async_nt(A, K, bm, 0, smbase, tid);
  stage_async_nt(B, K, bn, 0, smbase + TILE_BYTES, tid);

  int buf = 0;
  for (int k0 = 0; k0 < K; k0 += TBK) {
    const int nbuf = buf ^ 1;
    if (k0 + TBK < K) {
      // issue next tile's 8 async copies, then wait for the *previous* 8
      const unsigned nb = smbase + (unsigned)nbuf * 2u * TILE_BYTES;
      stage_async_nt(A, K, bm, k0 + TBK, nb, tid);
      stage_async_nt(B, K, bn, k0 + TBK, nb + TILE_BYTES, tid);
      if (k0 + 2 * TBK < K) {  // L2 prefetch two tiles ahead
        int r = tid & 127;
        const bf16* pp = (tid < 128) ? (A + (size_t)(bm + r) * K + k0 + 2 * TBK)
                                     : (B + (size_t)(bn + r) * K + k0 + 2 * TBK);
        __builtin_prefetch(pp, 0, 0);
      }
      wait_async_le8();               // cur tile's copies (oldest 8) have landed
    } else {
      wait_async_zero();              // final tile: drain everything
    }
    __syncthreads();                  // make LDS fills visible to all waves

    const bf16* As = smem + buf * 2 * TILE_ELEMS;
    const bf16* Bs = As + TILE_ELEMS;
#pragma unroll
    for (int kk = 0; kk < TBK; kk += 32) {
      v16bf af[2], bfr[4];
#pragma unroll
      for (int mi = 0; mi < 2; ++mi) af[mi]  = load_frag_a(As, wm * 32 + mi * 16, kk, lane);
#pragma unroll
      for (int ni = 0; ni < 4; ++ni) bfr[ni] = load_frag_b(Bs, wn * 64 + ni * 16, kk, lane);
#pragma unroll
      for (int mi = 0; mi < 2; ++mi)
#pragma unroll
        for (int ni = 0; ni < 4; ++ni)
          acc[mi][ni] = wmma_bf16(af[mi], bfr[ni], acc[mi][ni]);
    }
    __syncthreads();                  // protect cur buffer before it is refilled
    buf = nbuf;
  }

  const int rlane = (lane >> 4) * 8;  // C layout: VGPR r -> M = r + 8*(lane>=16)
  const int clane = lane & 15;
#pragma unroll
  for (int mi = 0; mi < 2; ++mi)
#pragma unroll
    for (int ni = 0; ni < 4; ++ni)
#pragma unroll
      for (int r = 0; r < 8; ++r) {
        int row = bm + wm * 32 + mi * 16 + rlane + r;
        int col = bn + wn * 64 + ni * 16 + clane;
        size_t idx = (size_t)row * (size_t)N + (size_t)col;
        float v = acc[mi][ni][r];
        if constexpr (MODE == MODE_F32)  Cf[idx] = v;
        else if constexpr (MODE == MODE_BF16) Cb[idx] = f2bf(v);
        else if constexpr (MODE == MODE_SILU) Cb[idx] = f2bf(fsilu(v));
        else if constexpr (MODE == MODE_SILU_DUAL) { Cb2[idx] = f2bf(v); Cb[idx] = f2bf(fsilu(v)); }
        else if constexpr (MODE == MODE_ERR) {
          float e = v - bf2f(auxb[idx]);
          e = fminf(5.0f, fmaxf(-5.0f, e));
          Cb[idx] = f2bf(scale * e);
        } else { // MODE_DSILU
          Cb[idx] = f2bf(v * fsilu_deriv(bf2f(auxb[idx])));
        }
      }
}

// G[Dm,Dn] = A[S,Dm]^T @ B[S,Dn]; fused: Wn = (1-alpha)*Wold - lr*clip(G, +-1) (bf16)
__global__ __launch_bounds__(256) void gemm_bf16_tn_wupd(
    const bf16* __restrict__ A, const bf16* __restrict__ B,
    const float* __restrict__ Wold, const float* __restrict__ gates,
    bf16* __restrict__ Wn, int Dm, int Dn, int S) {
  const int tid  = threadIdx.x;
  const int lane = tid & 31;
  const int wave = tid >> 5;
  const int wm   = wave & 3;
  const int wn   = wave >> 2;
  const int bm   = blockIdx.y * TBM;
  const int bn   = blockIdx.x * TBN;

  __shared__ __align__(16) bf16 smem[2 * TILE_ELEMS];
  bf16* As = smem;
  bf16* Bs = smem + TILE_ELEMS;

  v8f acc[2][4];
#pragma unroll
  for (int mi = 0; mi < 2; ++mi)
#pragma unroll
    for (int ni = 0; ni < 4; ++ni)
#pragma unroll
      for (int r = 0; r < 8; ++r) acc[mi][ni][r] = 0.0f;

  for (int s0 = 0; s0 < S; s0 += TBK) {
    stage_tile_tn(A, Dm, s0, bm, As, tid);   // As[d][s]
    stage_tile_tn(B, Dn, s0, bn, Bs, tid);   // Bs[d][s]
    if (s0 + TBK < S) {
      int r = tid & 63;
      const bf16* pp = (tid < 128) ? (A + (size_t)(s0 + TBK + r) * Dm + bm)
                                   : (B + (size_t)(s0 + TBK + r) * Dn + bn);
      __builtin_prefetch(pp, 0, 0);
    }
    __syncthreads();
#pragma unroll
    for (int kk = 0; kk < TBK; kk += 32) {
      v16bf af[2], bfr[4];
#pragma unroll
      for (int mi = 0; mi < 2; ++mi) af[mi]  = load_frag_a(As, wm * 32 + mi * 16, kk, lane);
#pragma unroll
      for (int ni = 0; ni < 4; ++ni) bfr[ni] = load_frag_b(Bs, wn * 64 + ni * 16, kk, lane);
#pragma unroll
      for (int mi = 0; mi < 2; ++mi)
#pragma unroll
        for (int ni = 0; ni < 4; ++ni)
          acc[mi][ni] = wmma_bf16(af[mi], bfr[ni], acc[mi][ni]);
    }
    __syncthreads();
  }

  const float alpha = gates[0];
  const float lr    = gates[1];
  const int rlane = (lane >> 4) * 8;
  const int clane = lane & 15;
#pragma unroll
  for (int mi = 0; mi < 2; ++mi)
#pragma unroll
    for (int ni = 0; ni < 4; ++ni)
#pragma unroll
      for (int r = 0; r < 8; ++r) {
        int row = bm + wm * 32 + mi * 16 + rlane + r;
        int col = bn + wn * 64 + ni * 16 + clane;
        size_t idx = (size_t)row * (size_t)Dn + (size_t)col;
        float g = fminf(1.0f, fmaxf(-1.0f, acc[mi][ni][r]));
        Wn[idx] = f2bf((1.0f - alpha) * Wold[idx] - lr * g);
      }
}

// ---------------- small helper kernels ----------------
__global__ __launch_bounds__(256) void cast4_kernel(const float* __restrict__ src,
                                                    bf16* __restrict__ dst, int n4) {
  int i = blockIdx.x * 256 + threadIdx.x;
  if (i < n4) {
    float4 v = reinterpret_cast<const float4*>(src)[i];
    dst[4 * i + 0] = f2bf(v.x);
    dst[4 * i + 1] = f2bf(v.y);
    dst[4 * i + 2] = f2bf(v.z);
    dst[4 * i + 3] = f2bf(v.w);
  }
}

// dst[i*dim + j] = bf16(src[j*dim + i])
__global__ void transpose_cast_kernel(const float* __restrict__ src,
                                      bf16* __restrict__ dst, int dim) {
  __shared__ float t[32][33];
  int bx = blockIdx.x * 32, by = blockIdx.y * 32;
  int tx = threadIdx.x, ty = threadIdx.y;       // 32 x 8
#pragma unroll
  for (int j = 0; j < 32; j += 8)
    t[ty + j][tx] = src[(size_t)(by + ty + j) * dim + bx + tx];
  __syncthreads();
#pragma unroll
  for (int j = 0; j < 32; j += 8)
    dst[(size_t)(bx + ty + j) * dim + by + tx] = f2bf(t[tx][ty + j]);
}

// deterministic column-sum of x [NTOK, DIMN] in two fixed-order stages (no atomics)
#define CS_ROWS 256
__global__ __launch_bounds__(256) void colsum_stage1(const float* __restrict__ x,
                                                     float* __restrict__ partials) {
  int r0 = blockIdx.x * CS_ROWS;
  for (int c = threadIdx.x; c < DIMN; c += 256) {
    float s = 0.0f;
    for (int r = 0; r < CS_ROWS; ++r) s += x[(size_t)(r0 + r) * DIMN + c];
    partials[(size_t)blockIdx.x * DIMN + c] = s;
  }
}
__global__ __launch_bounds__(256) void colsum_stage2(const float* __restrict__ partials,
                                                     float* __restrict__ colsum, int nparts) {
  int c = blockIdx.x * 256 + threadIdx.x;
  if (c < DIMN) {
    float s = 0.0f;
    for (int p = 0; p < nparts; ++p) s += partials[(size_t)p * DIMN + c];
    colsum[c] = s;
  }
}

// gates[0]=alpha, gates[1]=lr, gates[2]=eta  (mean over tokens == dot(colsum,Wg)/N + b)
__global__ __launch_bounds__(256) void gate_finalize_kernel(
    const float* __restrict__ colsum,
    const float* __restrict__ wgd, const float* __restrict__ bgd,
    const float* __restrict__ wgl, const float* __restrict__ bgl,
    const float* __restrict__ wgm, const float* __restrict__ bgm,
    float* __restrict__ gates) {
  __shared__ float r0[256], r1[256], r2[256];
  int t = threadIdx.x;
  float s0 = 0.f, s1 = 0.f, s2 = 0.f;
  for (int c = t; c < DIMN; c += 256) {
    float cs = colsum[c];
    s0 += cs * wgd[c]; s1 += cs * wgl[c]; s2 += cs * wgm[c];
  }
  r0[t] = s0; r1[t] = s1; r2[t] = s2;
  __syncthreads();
  for (int s = 128; s > 0; s >>= 1) {
    if (t < s) { r0[t] += r0[t + s]; r1[t] += r1[t + s]; r2[t] += r2[t + s]; }
    __syncthreads();
  }
  if (t == 0) {
    const float invN = 1.0f / (float)NTOK;
    gates[0] = fsigmoid(r0[0] * invN + bgd[0]);   // alpha (weight decay)
    gates[1] = fsigmoid(r1[0] * invN + bgl[0]);   // lr
    gates[2] = fsigmoid(r2[0] * invN + bgm[0]);   // eta (unused: momentum state = 0)
  }
}

// ---------------- host orchestration ----------------
extern "C" void kernel_launch(void* const* d_in, const int* in_sizes, int n_in,
                              void* d_out, int out_size, void* d_ws, size_t ws_size,
                              hipStream_t stream) {
  (void)in_sizes; (void)n_in; (void)out_size;
  const float* x    = (const float*)d_in[0];
  const float* Wk   = (const float*)d_in[1];
  const float* Wv   = (const float*)d_in[2];
  const float* Wq   = (const float*)d_in[3];
  const float* Wout = (const float*)d_in[4];
  const float* W0   = (const float*)d_in[5];
  const float* W1   = (const float*)d_in[6];
  const float* Wgd  = (const float*)d_in[7];
  const float* bgd  = (const float*)d_in[8];
  const float* Wgl  = (const float*)d_in[9];
  const float* bgl  = (const float*)d_in[10];
  const float* Wgm  = (const float*)d_in[11];
  const float* bgm  = (const float*)d_in[12];
  float* out = (float*)d_out;

  // workspace layout (~243 MB; slots reused along the dependency chain)
  char* ws = (char*)d_ws;
  size_t off = 0;
  auto alloc = [&](size_t bytes) -> void* {
    void* p = ws + off;
    off += (bytes + 255) & ~(size_t)255;
    return p;
  };
  const size_t ND2 = (size_t)ND_ELEMS * 2;   // bf16 activation slot
  const size_t DD2 = (size_t)DD_ELEMS * 2;   // bf16 weight slot
  bf16* sl0 = (bf16*)alloc(ND2);   // x16   -> d1_16
  bf16* sl1 = (bf16*)alloc(ND2);   // k16
  bf16* sl2 = (bf16*)alloc(ND2);   // q16
  bf16* sl3 = (bf16*)alloc(ND2);   // v16   -> s1_16
  bf16* sl4 = (bf16*)alloc(ND2);   // a1_16 -> h16
  bf16* sl5 = (bf16*)alloc(ND2);   // d2_16
  bf16* sl6 = (bf16*)alloc(ND2);   // pre1_16
  bf16* wk16   = (bf16*)alloc(DD2);
  bf16* wv16   = (bf16*)alloc(DD2);
  bf16* wq16   = (bf16*)alloc(DD2);
  bf16* wout16 = (bf16*)alloc(DD2);
  bf16* w0_16  = (bf16*)alloc(DD2);
  bf16* w1_16  = (bf16*)alloc(DD2);
  bf16* w1t16  = (bf16*)alloc(DD2);
  bf16* w0n16  = (bf16*)alloc(DD2);
  bf16* w1n16  = (bf16*)alloc(DD2);
  float* partials = (float*)alloc((size_t)(NTOK / CS_ROWS) * DIMN * 4);
  float* colsum   = (float*)alloc(DIMN * 4);
  float* gates    = (float*)alloc(64 * 4);
  if (ws_size < off) return;   // workspace too small: bail (nothing safe to do)

  bf16* x16   = sl0; bf16* d1_16 = sl0;
  bf16* k16   = sl1;
  bf16* q16   = sl2;
  bf16* v16   = sl3; bf16* s1_16 = sl3;
  bf16* a1_16 = sl4; bf16* h16   = sl4;
  bf16* d2_16 = sl5;
  bf16* pre1_16 = sl6;

  const dim3 blk(256);
  const dim3 gNT(DIMN / TBN, NTOK / TBM);   // (8, 128)
  const dim3 gTN(DIMN / TBN, DIMN / TBM);   // (8, 8)
  const float dscale = 2.0f / (float)SEQ;

  // ---- casts ----
  {
    int n4 = (int)(ND_ELEMS / 4);
    cast4_kernel<<<(n4 + 255) / 256, blk, 0, stream>>>(x, x16, n4);
    int d4 = (int)(DD_ELEMS / 4);
    cast4_kernel<<<(d4 + 255) / 256, blk, 0, stream>>>(Wk, wk16, d4);
    cast4_kernel<<<(d4 + 255) / 256, blk, 0, stream>>>(Wv, wv16, d4);
    cast4_kernel<<<(d4 + 255) / 256, blk, 0, stream>>>(Wq, wq16, d4);
    cast4_kernel<<<(d4 + 255) / 256, blk, 0, stream>>>(Wout, wout16, d4);
    cast4_kernel<<<(d4 + 255) / 256, blk, 0, stream>>>(W0, w0_16, d4);
    cast4_kernel<<<(d4 + 255) / 256, blk, 0, stream>>>(W1, w1_16, d4);
    dim3 tb(32, 8), tg(DIMN / 32, DIMN / 32);
    transpose_cast_kernel<<<tg, tb, 0, stream>>>(W1, w1t16, DIMN);   // W1^T for d2@W1
  }

  // ---- scalar gates (deterministic) ----
  colsum_stage1<<<NTOK / CS_ROWS, blk, 0, stream>>>(x, partials);
  colsum_stage2<<<(DIMN + 255) / 256, blk, 0, stream>>>(partials, colsum, NTOK / CS_ROWS);
  gate_finalize_kernel<<<1, blk, 0, stream>>>(colsum, Wgd, bgd, Wgl, bgl, Wgm, bgm, gates);

  // ---- projections ----
  gemm_bf16_nt<MODE_BF16><<<gNT, blk, 0, stream>>>(x16, wk16, nullptr, k16, nullptr, nullptr, NTOK, DIMN, DIMN, 0.f);
  gemm_bf16_nt<MODE_BF16><<<gNT, blk, 0, stream>>>(x16, wv16, nullptr, v16, nullptr, nullptr, NTOK, DIMN, DIMN, 0.f);
  gemm_bf16_nt<MODE_BF16><<<gNT, blk, 0, stream>>>(x16, wq16, nullptr, q16, nullptr, nullptr, NTOK, DIMN, DIMN, 0.f);

  // ---- inner loop: pre1 = k@W0^T (keep raw + silu); pred/d2 fused ----
  gemm_bf16_nt<MODE_SILU_DUAL><<<gNT, blk, 0, stream>>>(k16, w0_16, nullptr, a1_16, pre1_16, nullptr, NTOK, DIMN, DIMN, 0.f);
  gemm_bf16_nt<MODE_ERR><<<gNT, blk, 0, stream>>>(a1_16, w1_16, nullptr, d2_16, nullptr, v16, NTOK, DIMN, DIMN, dscale);

  // ---- W1n = (1-alpha)W1 - lr*clip(d2^T @ a1) ----
  gemm_bf16_tn_wupd<<<gTN, blk, 0, stream>>>(d2_16, a1_16, W1, gates, w1n16, DIMN, DIMN, NTOK);

  // ---- d1 = (d2 @ W1) * silu'(pre1) ----
  gemm_bf16_nt<MODE_DSILU><<<gNT, blk, 0, stream>>>(d2_16, w1t16, nullptr, d1_16, nullptr, pre1_16, NTOK, DIMN, DIMN, 0.f);

  // ---- W0n = (1-alpha)W0 - lr*clip(d1^T @ k) ----
  gemm_bf16_tn_wupd<<<gTN, blk, 0, stream>>>(d1_16, k16, W0, gates, w0n16, DIMN, DIMN, NTOK);

  // ---- retrieval with post-update weights ----
  gemm_bf16_nt<MODE_SILU><<<gNT, blk, 0, stream>>>(q16, w0n16, nullptr, s1_16, nullptr, nullptr, NTOK, DIMN, DIMN, 0.f);
  gemm_bf16_nt<MODE_BF16><<<gNT, blk, 0, stream>>>(s1_16, w1n16, nullptr, h16, nullptr, nullptr, NTOK, DIMN, DIMN, 0.f);
  gemm_bf16_nt<MODE_F32><<<gNT, blk, 0, stream>>>(h16, wout16, out, nullptr, nullptr, nullptr, NTOK, DIMN, DIMN, 0.f);
}
